// QKVAttentionLegacy_3942779977840
// MI455X (gfx1250) — compile-verified, matrix operand order checked by reading, not run on hardware
//
#include <hip/hip_runtime.h>

// QKV attention (QKVAttentionLegacy), flash-attention style, bf16 WMMA, wave32.
// Async b128 global->LDS double-buffered fp32 staging + one-shot LDS convert
// pass to bf16 tiles; DPP row_xmask softmax reductions; scale folded into Q.
// qkv: [N=8][3*H*C=1536][T=2048] fp32 ; out: [N][H*C=512][T] fp32.

#define T_LEN 2048
#define C_DIM 64
#define HEADS_TOTAL 64          // N*H
#define M_TILE 128              // query tokens per workgroup (16 per wave)
#define NWAVES 8
#define K_BLK 64                // key tokens per inner iteration
#define NBLK (T_LEN / K_BLK)
#define NTHREADS 256            // 8 wave32
#define FPITCH 68               // fp32 staging pitch   [channel][token]
#define BPITCH 68               // bf16 tile pitch (rows of 64 + 4 pad)

#define AS1 __attribute__((address_space(1)))
#define AS3 __attribute__((address_space(3)))
typedef int gvec4i __attribute__((vector_size(16)));   // matches builtin's param

#if __has_builtin(__builtin_amdgcn_global_load_async_to_lds_b128)
#define HAVE_ASYNC 1
#else
#define HAVE_ASYNC 0
#endif

#if __has_builtin(__builtin_amdgcn_s_wait_asynccnt)
#define WAIT_ASYNC() __builtin_amdgcn_s_wait_asynccnt(0)
#elif HAVE_ASYNC
#define WAIT_ASYNC() asm volatile("s_wait_asynccnt 0" ::: "memory")
#else
#define WAIT_ASYNC()
#endif

typedef __attribute__((ext_vector_type(16))) __bf16 v16bf;
typedef __attribute__((ext_vector_type(8)))  float  v8f;

union Frag16 {
  v16bf bf;
  unsigned int w[8];
};
union Frag32 {
  v8f f;
  float x[8];
};

__device__ __forceinline__ unsigned short f32_bf16(float f) {
  unsigned int u = __builtin_bit_cast(unsigned int, f);
  u += 0x7FFFu + ((u >> 16) & 1u);            // RNE
  return (unsigned short)(u >> 16);
}

// Round two floats to bf16 and pack into one dword via v_perm_b32.
__device__ __forceinline__ unsigned int pack_bf16(float x, float y) {
  unsigned int a = __builtin_bit_cast(unsigned int, x);
  unsigned int b = __builtin_bit_cast(unsigned int, y);
  a += 0x7FFFu + ((a >> 16) & 1u);
  b += 0x7FFFu + ((b >> 16) & 1u);
  return __builtin_amdgcn_perm(b, a, 0x07060302u);  // {b.hi16, a.hi16}
}

// XOR-butterfly within each 16-lane row via DPP16 ROW_XMASK (0x160|mask).
template <int MASK>
__device__ __forceinline__ float dpp_xor16(float x) {
  int v = __builtin_bit_cast(int, x);
  v = __builtin_amdgcn_update_dpp(0, v, 0x160 | MASK, 0xf, 0xf, true);
  return __builtin_bit_cast(float, v);
}
__device__ __forceinline__ float reduce16_max(float x) {
  x = fmaxf(x, dpp_xor16<1>(x));
  x = fmaxf(x, dpp_xor16<2>(x));
  x = fmaxf(x, dpp_xor16<4>(x));
  x = fmaxf(x, dpp_xor16<8>(x));
  return x;
}
__device__ __forceinline__ float reduce16_sum(float x) {
  x += dpp_xor16<1>(x);
  x += dpp_xor16<2>(x);
  x += dpp_xor16<4>(x);
  x += dpp_xor16<8>(x);
  return x;
}

// Load a 16x32 WMMA operand from bf16 LDS (row-major): lanes 0-15 rows 0-15
// K {0..7,16..23}; lanes 16-31 same rows, K {8..15,24..31}; 2 bf16 per dword.
__device__ __forceinline__ void load_frag_bf(const unsigned short* row_base,
                                             int pitch, int lane, Frag16& f) {
  const unsigned short* p = row_base + (lane & 15) * pitch + ((lane >> 4) << 3);
#pragma unroll
  for (int i = 0; i < 4; ++i) {
    f.w[i]     = *(const unsigned int*)(p + 2 * i);
    f.w[4 + i] = *(const unsigned int*)(p + 16 + 2 * i);
  }
}

// Async-stage one K/V block (fp32, natural [channel][token] layout, b128).
__device__ __forceinline__ void stage_kv(const float* kp, const float* vp,
                                         int s0, float* kf, float* vf, int tid) {
#if HAVE_ASYNC
  for (int i = tid; i < C_DIM * (K_BLK / 4); i += NTHREADS) {   // 4 iters
    int s4 = i & (K_BLK / 4 - 1);
    int c  = i >> 4;
    const AS1 float* gk = (const AS1 float*)(kp + (size_t)c * T_LEN + s0 + s4 * 4);
    const AS1 float* gv = (const AS1 float*)(vp + (size_t)c * T_LEN + s0 + s4 * 4);
    AS3 float* lk = (AS3 float*)(kf + c * FPITCH + s4 * 4);
    AS3 float* lv = (AS3 float*)(vf + c * FPITCH + s4 * 4);
    __builtin_amdgcn_global_load_async_to_lds_b128((AS1 gvec4i*)gk, (AS3 gvec4i*)lk, 0, 0);
    __builtin_amdgcn_global_load_async_to_lds_b128((AS1 gvec4i*)gv, (AS3 gvec4i*)lv, 0, 0);
  }
#else
  for (int i = tid; i < K_BLK * C_DIM; i += NTHREADS) {
    int s = i & (K_BLK - 1);
    int c = i >> 6;
    kf[c * FPITCH + s] = kp[(size_t)c * T_LEN + s0 + s];
    vf[c * FPITCH + s] = vp[(size_t)c * T_LEN + s0 + s];
  }
#endif
}

__global__ __launch_bounds__(NTHREADS)
void qkv_attn_kernel(const float* __restrict__ qkv, float* __restrict__ out) {
  __shared__ __align__(16) float          lds_kf[2][C_DIM][FPITCH];  // fp32 stage
  __shared__ __align__(16) float          lds_vf[2][C_DIM][FPITCH];
  __shared__ __align__(16) unsigned short lds_q [M_TILE][BPITCH];    // bf16 [t][c]
  __shared__ __align__(16) unsigned short lds_kb[K_BLK][BPITCH];     // bf16 [t][c]
  __shared__ __align__(16) unsigned short lds_vb[C_DIM][BPITCH];     // bf16 [c][t]
  __shared__ __align__(16) unsigned short lds_p [NWAVES][16][BPITCH];// bf16 P pads

  const int tid  = threadIdx.x;
  const int wave = tid >> 5;
  const int lane = tid & 31;
  const int bh   = blockIdx.x;
  const int t0   = blockIdx.y * M_TILE;

  const float* qp = qkv + (size_t)bh * 3 * C_DIM * T_LEN;
  const float* kp = qp + (size_t)C_DIM * T_LEN;
  const float* vp = kp + (size_t)C_DIM * T_LEN;

  // Kick off async staging of key-block 0 immediately.
  stage_kv(kp, vp, 0, &lds_kf[0][0][0], &lds_vf[0][0][0], tid);

  // Stage Q tile once, pre-scaled by scale^2*log2(e) so WMMA scores land
  // directly in the base-2 softmax domain (bf16 keeps fp32's exponent range).
  const float qscale = 1.4426950408889634f / (float)T_LEN;
  for (int i = tid; i < C_DIM * (M_TILE / 4); i += NTHREADS) {   // 8 iters
    int tq = i & (M_TILE / 4 - 1);
    int c  = i >> 5;
    float4 q4 = *(const float4*)(qp + (size_t)c * T_LEN + t0 + tq * 4);
    lds_q[tq * 4 + 0][c] = f32_bf16(q4.x * qscale);
    lds_q[tq * 4 + 1][c] = f32_bf16(q4.y * qscale);
    lds_q[tq * 4 + 2][c] = f32_bf16(q4.z * qscale);
    lds_q[tq * 4 + 3][c] = f32_bf16(q4.w * qscale);
  }
  __syncthreads();

  // Hoist loop-invariant Q fragments (two 32-channel k-steps).
  Frag16 aq0, aq1;
  load_frag_bf(&lds_q[wave * 16][0],  BPITCH, lane, aq0);
  load_frag_bf(&lds_q[wave * 16][32], BPITCH, lane, aq1);

  Frag32 o[4];
  float m_run[8], l_run[8];
#pragma unroll
  for (int f = 0; f < 4; ++f)
#pragma unroll
    for (int j = 0; j < 8; ++j) o[f].x[j] = 0.0f;
#pragma unroll
  for (int j = 0; j < 8; ++j) { m_run[j] = -1.0e30f; l_run[j] = 0.0f; }

  const v8f zero8 = {0.f, 0.f, 0.f, 0.f, 0.f, 0.f, 0.f, 0.f};

  int buf = 0;
  for (int kb = 0; kb < NBLK; ++kb) {
    WAIT_ASYNC();        // this wave's async fills of `buf` are complete
    __syncthreads();     // fills visible everywhere; bf16 tiles no longer read
    if (kb + 1 < NBLK)
      stage_kv(kp, vp, (kb + 1) * K_BLK,
               &lds_kf[buf ^ 1][0][0], &lds_vf[buf ^ 1][0][0], tid);

    // ---- Convert pass: fp32 staging -> bf16 tiles (each element once).
    {
      const float* kf = &lds_kf[buf][0][0];
      const float* vf = &lds_vf[buf][0][0];
      for (int i = tid; i < C_DIM * (K_BLK / 2); i += NTHREADS) {  // 8 iters
        int sp = i & (K_BLK / 2 - 1);          // token pair
        int c  = i >> 5;
        float2 kk = *(const float2*)(kf + c * FPITCH + sp * 2);
        float2 vv = *(const float2*)(vf + c * FPITCH + sp * 2);
        lds_kb[sp * 2 + 0][c] = f32_bf16(kk.x);          // transpose K
        lds_kb[sp * 2 + 1][c] = f32_bf16(kk.y);
        *(unsigned int*)&lds_vb[c][sp * 2] = pack_bf16(vv.x, vv.y);
      }
    }
    __syncthreads();     // bf16 tiles ready

    // ---- GEMM1: S = Q K^T over C=64 (4 key sub-tiles of 16).
    Frag32 s2[4];
#pragma unroll
    for (int blk = 0; blk < 4; ++blk) {
      Frag16 b0, b1;
      load_frag_bf(&lds_kb[blk * 16][0],  BPITCH, lane, b0);
      load_frag_bf(&lds_kb[blk * 16][32], BPITCH, lane, b1);
      v8f acc = zero8;
      acc = __builtin_amdgcn_wmma_f32_16x16x32_bf16(false, aq0.bf, false, b0.bf,
                                                    (short)0, acc, false, false);
      acc = __builtin_amdgcn_wmma_f32_16x16x32_bf16(false, aq1.bf, false, b1.bf,
                                                    (short)0, acc, false, false);
      s2[blk].f = acc;
    }

    // ---- Online softmax over the 64 new columns (base-2 domain).
    float rmax[8], alpha[8], rsum[8];
#pragma unroll
    for (int j = 0; j < 8; ++j)
      rmax[j] = fmaxf(fmaxf(s2[0].x[j], s2[1].x[j]),
                      fmaxf(s2[2].x[j], s2[3].x[j]));
#pragma unroll
    for (int j = 0; j < 8; ++j) rmax[j] = reduce16_max(rmax[j]);
#pragma unroll
    for (int j = 0; j < 8; ++j) {
      float mnew = fmaxf(m_run[j], rmax[j]);
      alpha[j]   = __builtin_amdgcn_exp2f(m_run[j] - mnew);
      m_run[j]   = mnew;
      s2[0].x[j] = __builtin_amdgcn_exp2f(s2[0].x[j] - mnew);
      s2[1].x[j] = __builtin_amdgcn_exp2f(s2[1].x[j] - mnew);
      s2[2].x[j] = __builtin_amdgcn_exp2f(s2[2].x[j] - mnew);
      s2[3].x[j] = __builtin_amdgcn_exp2f(s2[3].x[j] - mnew);
      rsum[j] = (s2[0].x[j] + s2[1].x[j]) + (s2[2].x[j] + s2[3].x[j]);
    }
#pragma unroll
    for (int j = 0; j < 8; ++j) rsum[j] = reduce16_sum(rsum[j]);
#pragma unroll
    for (int j = 0; j < 8; ++j) l_run[j] = l_run[j] * alpha[j] + rsum[j];
#pragma unroll
    for (int f = 0; f < 4; ++f)
#pragma unroll
      for (int j = 0; j < 8; ++j) o[f].x[j] *= alpha[j];

    // ---- P (C/D layout) -> per-wave LDS pad (bf16) -> A layout.
    {
      const int prow_base = (lane >> 4) << 3;
      const int pcol      = lane & 15;
#pragma unroll
      for (int blk = 0; blk < 4; ++blk)
#pragma unroll
        for (int j = 0; j < 8; ++j)
          lds_p[wave][prow_base + j][blk * 16 + pcol] = f32_bf16(s2[blk].x[j]);
    }

    // ---- GEMM2: O += P V^T  (K = 64 keys in two k-steps, 4 channel groups).
    Frag16 pa0, pa1;
    load_frag_bf(&lds_p[wave][0][0],  BPITCH, lane, pa0);
    load_frag_bf(&lds_p[wave][0][32], BPITCH, lane, pa1);
#pragma unroll
    for (int f = 0; f < 4; ++f) {
      Frag16 vb0, vb1;
      load_frag_bf(&lds_vb[f * 16][0],  BPITCH, lane, vb0);
      load_frag_bf(&lds_vb[f * 16][32], BPITCH, lane, vb1);
      o[f].f = __builtin_amdgcn_wmma_f32_16x16x32_bf16(false, pa0.bf, false, vb0.bf,
                                                       (short)0, o[f].f, false, false);
      o[f].f = __builtin_amdgcn_wmma_f32_16x16x32_bf16(false, pa1.bf, false, vb1.bf,
                                                       (short)0, o[f].f, false, false);
    }
    buf ^= 1;
  }

  // ---- Epilogue: normalize; rows of O are 8 consecutive tokens -> b128 stores.
  float invl[8];
#pragma unroll
  for (int j = 0; j < 8; ++j) invl[j] = 1.0f / l_run[j];
  const int tok_base = t0 + wave * 16 + ((lane >> 4) << 3);
#pragma unroll
  for (int f = 0; f < 4; ++f) {
    int c = f * 16 + (lane & 15);
    float* op = out + ((size_t)bh * C_DIM + c) * T_LEN + tok_base;
    float4 lo = {o[f].x[0] * invl[0], o[f].x[1] * invl[1],
                 o[f].x[2] * invl[2], o[f].x[3] * invl[3]};
    float4 hi = {o[f].x[4] * invl[4], o[f].x[5] * invl[5],
                 o[f].x[6] * invl[6], o[f].x[7] * invl[7]};
    *(float4*)(op)     = lo;
    *(float4*)(op + 4) = hi;
  }
}

extern "C" void kernel_launch(void* const* d_in, const int* in_sizes, int n_in,
                              void* d_out, int out_size, void* d_ws, size_t ws_size,
                              hipStream_t stream) {
  (void)in_sizes; (void)n_in; (void)out_size; (void)d_ws; (void)ws_size;
  const float* qkv = (const float*)d_in[0];   // d_in[1] = n_heads (compile-time 8)
  float* out = (float*)d_out;
  dim3 grid(HEADS_TOTAL, T_LEN / M_TILE, 1);
  dim3 block(NTHREADS, 1, 1);
  hipLaunchKernelGGL(qkv_attn_kernel, grid, block, 0, stream, qkv, out);
}
